// SelfAttention_26688926777714
// MI455X (gfx1250) — matrix operand order repaired and checked by reading, not compiled, not run on hardware
//
#include <hip/hip_runtime.h>
#include <hip/hip_bf16.h>

#define N_SP   4096   // H*W
#define C_DIM  256
#define MID_DIM 32
#define BATCH  8

typedef __attribute__((ext_vector_type(16))) __bf16 bf16x16;
typedef __attribute__((ext_vector_type(8)))  float  f32x8;

// 16-bit A-operand (16xK=32) element->K map (matches B):
// lanes 0-15: K = 0..15; lanes 16-31: K = 16..31
__device__ __forceinline__ int amapK(int e, int g) {
    return e + (g << 4);
}
// 16-bit B-operand (K=32 x 16) element->K map:
// lanes 0-15: K = 0..15; lanes 16-31: K = 16..31
__device__ __forceinline__ int bmapK(int e, int g) {
    return e + (g << 4);
}

__device__ __forceinline__ f32x8 wmma_bf16(bf16x16 a, bf16x16 b, f32x8 c) {
    return __builtin_amdgcn_wmma_f32_16x16x32_bf16(false, a, false, b, (short)0, c, false, false);
}

// ---------------------------------------------------------------------------
// 1x1 conv projection.
//   NMAJOR=true : dst[b][n][o] (n-major, odim=32)  -> contiguous WMMA operand
//                 fetches for the logits GEMMs, contiguous 16B stores.
//   NMAJOR=false: dst[b][o][n] (k-major, odim=256) -> contiguous B-operand
//                 fetches for the output GEMM.
// grid: (N/16, odim/16, B), block 32 (one wave)
// ---------------------------------------------------------------------------
template <bool NMAJOR>
__global__ void proj_kernel(const float* __restrict__ x, const float* __restrict__ w,
                            const float* __restrict__ bias, __bf16* __restrict__ dst,
                            int odim) {
    const int lane = threadIdx.x;
    const int col  = lane & 15;
    const int g    = lane >> 4;
    const int b  = blockIdx.z;
    const int ot = blockIdx.y;
    const int nt = blockIdx.x;
    const int nbase = nt * 16;
    const float* xb = x + (size_t)b * C_DIM * N_SP;

    f32x8 acc = {};
    for (int k0 = 0; k0 < C_DIM; k0 += 32) {
        bf16x16 aop, bop;
#pragma unroll
        for (int e = 0; e < 16; ++e) {
            aop[e] = (__bf16)w[(ot * 16 + col) * C_DIM + k0 + amapK(e, g)];
            // coalesced across lanes: 16 lanes x 4B contiguous per (e,g)
            bop[e] = (__bf16)xb[(size_t)(k0 + bmapK(e, g)) * N_SP + nbase + col];
        }
        acc = wmma_bf16(aop, bop, acc);
    }
    __bf16* db = dst + (size_t)b * odim * N_SP;
    if (NMAJOR) {
        // lane writes 8 consecutive bf16: o = ot*16 + 8g + v
#pragma unroll
        for (int v = 0; v < 8; ++v) {
            const int o = ot * 16 + (g << 3) + v;
            db[(size_t)(nbase + col) * odim + o] = (__bf16)(acc[v] + bias[o]);
        }
    } else {
#pragma unroll
        for (int v = 0; v < 8; ++v) {
            const int o = ot * 16 + (g << 3) + v;
            db[(size_t)o * N_SP + nbase + col] = (__bf16)(acc[v] + bias[o]);
        }
    }
}

// ---------------------------------------------------------------------------
// Column softmax stats over i: m[j] = max_i s[i,j], Z[j] = sum_i exp(s-m)
// s[i,j] = sum_m A[m,i]*Bm[m,j]   (single K=32 WMMA per 16x16 tile)
// Abuf/Bbuf are n-major [N][32]: operand fetches are 2x b128 per lane.
// grid: (N/16, B), block 32
// ---------------------------------------------------------------------------
__global__ void stats_kernel(const __bf16* __restrict__ Ab, const __bf16* __restrict__ Bb,
                             float* __restrict__ mbuf, float* __restrict__ zbuf) {
    const int lane = threadIdx.x;
    const int col  = lane & 15;
    const int g    = lane >> 4;
    const int b  = blockIdx.y;
    const int jt = blockIdx.x;
    const __bf16* Abat = Ab + (size_t)b * MID_DIM * N_SP;
    const __bf16* Bbat = Bb + (size_t)b * MID_DIM * N_SP;

    bf16x16 bop;   // (K=m, N=j): 16 consecutive bf16 at row j, offset 16g
#pragma unroll
    for (int e = 0; e < 16; ++e)
        bop[e] = Bbat[(size_t)(jt * 16 + col) * MID_DIM + bmapK(e, g)];

    float runM = -3.0e38f, runZ = 0.0f;
    for (int it = 0; it < N_SP / 16; ++it) {
        bf16x16 aop;   // (M=i, K=m): 16 consecutive bf16 at row i, offset 16g
#pragma unroll
        for (int e = 0; e < 16; ++e)
            aop[e] = Abat[(size_t)(it * 16 + col) * MID_DIM + amapK(e, g)];
        f32x8 s = {};
        s = wmma_bf16(aop, bop, s);
        // D layout: this lane holds column j=jt*16+col, rows i = it*16 + v + 8g.
        float tmax = s[0];
#pragma unroll
        for (int v = 1; v < 8; ++v) tmax = fmaxf(tmax, s[v]);
        tmax = fmaxf(tmax, __shfl_xor(tmax, 16, 32));   // combine lane halves
        float newM = fmaxf(runM, tmax);
        float ls = 0.0f;
#pragma unroll
        for (int v = 0; v < 8; ++v) ls += __expf(s[v] - newM);
        ls += __shfl_xor(ls, 16, 32);
        runZ = runZ * __expf(runM - newM) + ls;
        runM = newM;
    }
    if (g == 0) {
        mbuf[b * N_SP + jt * 16 + col] = runM;
        zbuf[b * N_SP + jt * 16 + col] = runZ;
    }
}

// ---------------------------------------------------------------------------
// out[b][i][k] = sum_j softmax(s)[i,j] * c[k,j], P never materialized.
// Each wave: one 16-row i-tile x 64 k-columns, loops j in steps of 32.
// grid: (4, N/16, B), block 32, 1KB LDS for the s->A-operand transpose.
// ---------------------------------------------------------------------------
__global__ void attn_out_kernel(const __bf16* __restrict__ Ab, const __bf16* __restrict__ Bb,
                                const __bf16* __restrict__ Cb,
                                const float* __restrict__ mbuf, const float* __restrict__ zbuf,
                                float* __restrict__ outbuf) {
    __shared__ __bf16 p_sh[16][32];
    const int lane = threadIdx.x;
    const int col  = lane & 15;
    const int g    = lane >> 4;
    const int b  = blockIdx.z;
    const int it = blockIdx.y;
    const int kg = blockIdx.x;   // 64 output channels per wave
    const __bf16* Abat = Ab + (size_t)b * MID_DIM * N_SP;  // n-major [N][32]
    const __bf16* Bbat = Bb + (size_t)b * MID_DIM * N_SP;  // n-major [N][32]
    const __bf16* Cbat = Cb + (size_t)b * C_DIM * N_SP;    // k-major [256][N]
    const float* mb = mbuf + b * N_SP;
    const float* zb = zbuf + b * N_SP;

    bf16x16 aop;                              // A^T tile, invariant over j
#pragma unroll
    for (int e = 0; e < 16; ++e)
        aop[e] = Abat[(size_t)(it * 16 + col) * MID_DIM + amapK(e, g)];

    f32x8 acc[4] = {};

    for (int jb = 0; jb < N_SP; jb += 32) {
        // two 16-wide logits tiles -> normalized probabilities in LDS
#pragma unroll
        for (int h = 0; h < 2; ++h) {
            const int j0 = jb + h * 16;
            bf16x16 bop;                      // contiguous 32B per lane
#pragma unroll
            for (int e = 0; e < 16; ++e)
                bop[e] = Bbat[(size_t)(j0 + col) * MID_DIM + bmapK(e, g)];
            f32x8 s = {};
            s = wmma_bf16(aop, bop, s);
            const int j = j0 + col;
            const float mj   = mb[j];
            const float zinv = 1.0f / zb[j];
#pragma unroll
            for (int v = 0; v < 8; ++v)
                p_sh[v + (g << 3)][h * 16 + col] = (__bf16)(__expf(s[v] - mj) * zinv);
        }
        __syncthreads();
        bf16x16 pop;                          // P tile as A-operand (M=i, K=j)
#pragma unroll
        for (int e = 0; e < 16; ++e)
            pop[e] = p_sh[col][amapK(e, g)];
#pragma unroll
        for (int kt = 0; kt < 4; ++kt) {
            const int kbase = kg * 64 + kt * 16;
            bf16x16 cop;                      // C^T tile (K=j, N=k): contiguous
#pragma unroll
            for (int e = 0; e < 16; ++e)
                cop[e] = Cbat[(size_t)(kbase + col) * N_SP + jb + bmapK(e, g)];
            acc[kt] = wmma_bf16(pop, cop, acc[kt]);
        }
        __syncthreads();
    }

    float* ob = outbuf + (size_t)b * N_SP * C_DIM;   // [N][C] row-major
#pragma unroll
    for (int kt = 0; kt < 4; ++kt) {
#pragma unroll
        for (int v = 0; v < 8; ++v) {
            const int i = it * 16 + v + (g << 3);
            const int k = kg * 64 + kt * 16 + col;
            ob[(size_t)i * C_DIM + k] = acc[kt][v];
        }
    }
}

// ---------------------------------------------------------------------------
// Final conv: y = wo @ (beta*reinterpret(out) + x) + bo.
// The [B,N,C]->[B,C,N] raw view is a flat linear read of outbuf.
// grid: (N/16, C/16, B), block 32
// ---------------------------------------------------------------------------
__global__ void final_kernel(const float* __restrict__ x, const float* __restrict__ outbuf,
                             const float* __restrict__ wo, const float* __restrict__ bo,
                             const float* __restrict__ beta, float* __restrict__ y) {
    const int lane = threadIdx.x;
    const int col  = lane & 15;
    const int g    = lane >> 4;
    const int b  = blockIdx.z;
    const int ot = blockIdx.y;
    const int nt = blockIdx.x;
    const int nbase = nt * 16;
    const float bet = beta[0];
    const float* xb = x      + (size_t)b * C_DIM * N_SP;
    const float* fb = outbuf + (size_t)b * N_SP * C_DIM;  // flat view [C][N]

    f32x8 acc = {};
    for (int k0 = 0; k0 < C_DIM; k0 += 32) {
        bf16x16 aop, bop;
#pragma unroll
        for (int e = 0; e < 16; ++e) {
            aop[e] = (__bf16)wo[(ot * 16 + col) * C_DIM + k0 + amapK(e, g)];
            const size_t idx = (size_t)(k0 + bmapK(e, g)) * N_SP + nbase + col;
            bop[e] = (__bf16)(bet * fb[idx] + xb[idx]);
        }
        acc = wmma_bf16(aop, bop, acc);
    }
#pragma unroll
    for (int v = 0; v < 8; ++v) {
        const int o = ot * 16 + v + (g << 3);
        y[((size_t)b * C_DIM + o) * N_SP + nbase + col] = acc[v] + bo[o];
    }
}

// ---------------------------------------------------------------------------
extern "C" void kernel_launch(void* const* d_in, const int* in_sizes, int n_in,
                              void* d_out, int out_size, void* d_ws, size_t ws_size,
                              hipStream_t stream) {
    const float* x    = (const float*)d_in[0];
    const float* wa   = (const float*)d_in[1];
    const float* ba   = (const float*)d_in[2];
    const float* wb   = (const float*)d_in[3];
    const float* bb   = (const float*)d_in[4];
    const float* wc   = (const float*)d_in[5];
    const float* bc   = (const float*)d_in[6];
    const float* wo   = (const float*)d_in[7];
    const float* bo   = (const float*)d_in[8];
    const float* beta = (const float*)d_in[9];
    float* y = (float*)d_out;

    char* ws = (char*)d_ws;
    __bf16* Abuf = (__bf16*)(ws);                                    //  2 MB  [B][N][32]
    __bf16* Bbuf = (__bf16*)(ws + (size_t)2  * 1024 * 1024);         //  2 MB  [B][N][32]
    __bf16* Cbuf = (__bf16*)(ws + (size_t)4  * 1024 * 1024);         // 16 MB  [B][256][N]
    float*  mbuf = (float*) (ws + (size_t)20 * 1024 * 1024);         // 128 KB
    float*  zbuf = (float*) (ws + (size_t)20 * 1024 * 1024 + 131072);// 128 KB
    float*  obuf = (float*) (ws + (size_t)20 * 1024 * 1024 + 262144);// 32 MB  [B][N][256]

    dim3 blk(32, 1, 1);

    proj_kernel<true ><<<dim3(N_SP / 16, MID_DIM / 16, BATCH), blk, 0, stream>>>(x, wa, ba, Abuf, MID_DIM);
    proj_kernel<true ><<<dim3(N_SP / 16, MID_DIM / 16, BATCH), blk, 0, stream>>>(x, wb, bb, Bbuf, MID_DIM);
    proj_kernel<false><<<dim3(N_SP / 16, C_DIM  / 16, BATCH), blk, 0, stream>>>(x, wc, bc, Cbuf, C_DIM);

    stats_kernel<<<dim3(N_SP / 16, BATCH, 1), blk, 0, stream>>>(Abuf, Bbuf, mbuf, zbuf);

    attn_out_kernel<<<dim3(4, N_SP / 16, BATCH), blk, 0, stream>>>(Abuf, Bbuf, Cbuf, mbuf, zbuf, obuf);

    final_kernel<<<dim3(N_SP / 16, C_DIM / 16, BATCH), blk, 0, stream>>>(x, obuf, wo, bo, beta, y);
}
